// RNNDecoder_52115133170044
// MI455X (gfx1250) — matrix operand order repaired and checked by reading, not compiled, not run on hardware
//
#include <hip/hip_runtime.h>
#include <hip/hip_bf16.h>

// ---------------------------------------------------------------------------
// Graves handwriting RNN decoder for MI455X (gfx1250, wave32, WMMA).
// Persistent kernel: 16 blocks (one per 16-row batch tile), 512 threads
// (16 waves). Each wave owns 32 hidden units and computes their i/f/g/o gate
// columns with v_wmma_f32_16x16x32_bf16; cell state stays in registers for
// all 1024 timesteps.
// ---------------------------------------------------------------------------

typedef __attribute__((ext_vector_type(16))) __bf16 v16bf;
typedef __attribute__((ext_vector_type(8)))  float  v8f;

#define BB   256
#define TT   1024
#define UU   1024
#define EE   64
#define HID  512
#define GAT  2048          // 4*HID
#define KMIX 10
#define NOUT 121           // 6*M+1
#define KXW  96            // padded [x(3)|w(64)] -> 96 (3 k-blocks of 32)
#define KFC  1536          // [h0|h1|h2]

// workspace layout (bf16 element offsets)
constexpr size_t O_WXW0 = 0;                      // 2048 x 96
constexpr size_t O_WHH0 = O_WXW0 + 2048*96;       // 2048 x 512
constexpr size_t O_WXW1 = O_WHH0 + 2048*512;      // 2048 x 96
constexpr size_t O_WH01 = O_WXW1 + 2048*96;       // 2048 x 512   (h0 -> gates1)
constexpr size_t O_WHH1 = O_WH01 + 2048*512;      // 2048 x 512
constexpr size_t O_WXW2 = O_WHH1 + 2048*512;      // 2048 x 96
constexpr size_t O_WH12 = O_WXW2 + 2048*96;       // 2048 x 512   (h1 -> gates2)
constexpr size_t O_WHH2 = O_WH12 + 2048*512;      // 2048 x 512
constexpr size_t O_WFC  = O_WHH2 + 2048*512;      // 128  x 1536
constexpr size_t WS_TOT = O_WFC  + 128*1536;

// d_out layout (float element offsets), reference return order
constexpr size_t OFF_OUT = 0;
constexpr size_t OFF_PHI = OFF_OUT + (size_t)BB*TT*NOUT;
constexpr size_t OFF_KAP = OFF_PHI + (size_t)BB*TT*UU;
constexpr size_t OFF_ALP = OFF_KAP + (size_t)BB*TT*KMIX;
constexpr size_t OFF_BET = OFF_ALP + (size_t)BB*TT*KMIX;
constexpr size_t OFF_H0  = OFF_BET + (size_t)BB*TT*KMIX;
constexpr size_t OFF_C0  = OFF_H0  + (size_t)BB*HID;
constexpr size_t OFF_H1  = OFF_C0  + (size_t)BB*HID;
constexpr size_t OFF_C1  = OFF_H1  + (size_t)BB*HID;
constexpr size_t OFF_H2  = OFF_C1  + (size_t)BB*HID;
constexpr size_t OFF_C2  = OFF_H2  + (size_t)BB*HID;
constexpr size_t OFF_W   = OFF_C2  + (size_t)BB*HID;
constexpr size_t OFF_KC  = OFF_W   + (size_t)BB*EE;

// ---------------- fast math (v_exp_f32 / v_rcp_f32) ----------------
__device__ __forceinline__ float fexp(float x) {
  return __builtin_amdgcn_exp2f(x * 1.44269504088896340736f);
}
__device__ __forceinline__ float frcp(float x) { return __builtin_amdgcn_rcpf(x); }
__device__ __forceinline__ float fsig(float x) { return frcp(1.0f + fexp(-x)); }
__device__ __forceinline__ float ftanh(float x) {
  return 1.0f - 2.0f * frcp(fexp(2.0f * x) + 1.0f);
}

#define WMMA_BF16(A, B, C) \
  __builtin_amdgcn_wmma_f32_16x16x32_bf16(false, (A), false, (B), (short)0, (C), false, false)

// A fragment (16x32 bf16) from an LDS row-major activation tile.
// Contiguous k order (same as load_b): lane<16 -> M=lane, K in {0..15}; lane>=16 -> M=lane-16,
// K in {16..31}; 2 consecutive K per VGPR -> 2-bf16 (b32) LDS loads.
__device__ __forceinline__ v16bf load_a(const __bf16* row, int kbase, int hi) {
  v16bf a;
#pragma unroll
  for (int r = 0; r < 8; ++r) {
    int k = kbase + hi * 16 + (r << 1);
    a[2 * r]     = row[k];
    a[2 * r + 1] = row[k + 1];
  }
  return a;
}

// B fragment (32x16 bf16): lane<16 -> col N=lane, K=kbase..kbase+15;
// lane>=16 -> col N=lane-16, K=kbase+16..kbase+31. Weights kept in the
// original [N=4H rows, K cols] row-major order -> 16 contiguous bf16 = 32B.
__device__ __forceinline__ v16bf load_b(const __bf16* w, int ldk, int ncol,
                                        int kbase, int lane) {
  int n = lane & 15, hi = lane >> 4;
  return *(const v16bf*)(w + (size_t)(ncol + n) * ldk + kbase + hi * 16);
}

// accumulate [16 x 32*NKB] @ [32*NKB x 8*16] into 8 tiles
template <int NKB>
__device__ __forceinline__ void gemm_acc(v8f acc[8], const __bf16* act,
                                         int act_stride, const __bf16* w,
                                         int ldk, const int* ntile, int lane) {
  const __bf16* arow = act + (size_t)(lane & 15) * act_stride;
  int hi = lane >> 4;
#pragma unroll 1
  for (int kb = 0; kb < NKB; ++kb) {
    v16bf a = load_a(arow, kb * 32, hi);
#pragma unroll
    for (int t = 0; t < 8; ++t) {
      v16bf b = load_b(w, ldk, ntile[t], kb * 32, lane);
      acc[t] = WMMA_BF16(a, b, acc[t]);
    }
  }
}

__device__ __forceinline__ void init_gates(v8f acc[8], const float* bias,
                                           const int* ntile, int lane) {
#pragma unroll
  for (int t = 0; t < 8; ++t) {
    float bv = bias[ntile[t] + (lane & 15)];
#pragma unroll
    for (int r = 0; r < 8; ++r) acc[t][r] = bv;
  }
}

// tiles 0,1=i  2,3=f  4,5=g  6,7=o ; store h (bf16) into s_h[:,hbase+...]
__device__ __forceinline__ void lstm_update(v8f acc[8], v8f cst[2], __bf16* s_h,
                                            int hbase, int n0, int lane) {
  int col0 = n0 + (lane & 15);
  int hi = lane >> 4;
#pragma unroll
  for (int j = 0; j < 2; ++j) {
#pragma unroll
    for (int r = 0; r < 8; ++r) {
      float iv = fsig(acc[j][r]);
      float fv = fsig(acc[2 + j][r]);
      float gv = ftanh(acc[4 + j][r]);
      float ov = fsig(acc[6 + j][r]);
      float c  = fv * cst[j][r] + iv * gv;
      cst[j][r] = c;
      float h  = ov * ftanh(c);
      int m = r + hi * 8;
      s_h[(size_t)m * KFC + hbase + col0 + j * 16] = (__bf16)h;
    }
  }
}

// ---------------- weight conversion kernels ----------------
__global__ void k_zero_bf16(__bf16* p, int n) {
  int i = blockIdx.x * blockDim.x + threadIdx.x;
  if (i < n) p[i] = (__bf16)0.0f;
}

__global__ void k_cvt(const float* src, __bf16* dst, int rows, int scols,
                      int soff, int copyw, int dcols, int dcoff) {
  int i = blockIdx.x * blockDim.x + threadIdx.x;
  if (i >= rows * copyw) return;
  int r = i / copyw, c = i - r * copyw;
  dst[(size_t)r * dcols + dcoff + c] = (__bf16)src[(size_t)r * scols + soff + c];
}

// ---------------- persistent RNN kernel ----------------
__global__ __launch_bounds__(512, 1) void k_rnn(
    const float* __restrict__ strokes, const float* __restrict__ context,
    const float* __restrict__ b0, const float* __restrict__ b1,
    const float* __restrict__ b2, const float* __restrict__ Wa,
    const float* __restrict__ ba, const float* __restrict__ bfc,
    const __bf16* __restrict__ ws, float* __restrict__ out) {
  __shared__ __bf16 s_h[16 * KFC];    // [m][h0|h1|h2] bf16 (48 KB)
  __shared__ __bf16 s_xw[16 * KXW];   // [m][x(3)|w(64)|pad] bf16
  __shared__ float  s_wf[16 * EE];    // window vector f32 (carry)
  __shared__ float  s_kap[16 * KMIX];
  __shared__ float  s_alp[16 * KMIX];
  __shared__ float  s_bet[16 * KMIX];

  const int tid = threadIdx.x, lane = tid & 31, wv = tid >> 5;
  const int tile = blockIdx.x, brow0 = tile * 16;
  const int n0 = wv * 32;
  int ntile[8];
#pragma unroll
  for (int p = 0; p < 4; ++p) { ntile[2*p] = p * HID + n0; ntile[2*p+1] = p * HID + n0 + 16; }

  for (int i = tid; i < 16 * KFC; i += 512) s_h[i]  = (__bf16)0.0f;
  for (int i = tid; i < 16 * KXW; i += 512) s_xw[i] = (__bf16)0.0f;
  for (int i = tid; i < 16 * EE;  i += 512) s_wf[i] = 0.0f;
  if (tid < 16 * KMIX) s_kap[tid] = 0.0f;

  v8f c0st[2], c1st[2], c2st[2];
#pragma unroll
  for (int j = 0; j < 2; ++j)
#pragma unroll
    for (int r = 0; r < 8; ++r) { c0st[j][r] = 0.f; c1st[j][r] = 0.f; c2st[j][r] = 0.f; }

  __syncthreads();

  for (int t = 0; t < TT; ++t) {
    // ---- stage inputs: x_t and previous window w ----
    if (tid < 48) {
      int m = tid / 3, j = tid - 3 * m;
      s_xw[m * KXW + j] = (__bf16)strokes[((size_t)(brow0 + m) * TT + t) * 3 + j];
    }
    for (int i = tid; i < 16 * EE; i += 512) {
      int m = i >> 6, e = i & 63;
      s_xw[m * KXW + 3 + e] = (__bf16)s_wf[i];
    }
    __syncthreads();

    // ---- LSTM0 gates: [x|w] @ Wxw0 + h0_prev @ Whh0 ----
    v8f acc[8];
    init_gates(acc, b0, ntile, lane);
    gemm_acc<KXW / 32>(acc, s_xw, KXW, ws + O_WXW0, KXW, ntile, lane);
    gemm_acc<HID / 32>(acc, s_h, KFC, ws + O_WHH0, HID, ntile, lane);
    __syncthreads();                       // all reads of h0_prev done
    lstm_update(acc, c0st, s_h, 0, n0, lane);
    __syncthreads();

    // ---- attention mixture: abk = exp(h0 @ Wa.T + ba) ----
    if (tid < 16 * 3 * KMIX) {
      int m = tid / (3 * KMIX), j = tid - m * (3 * KMIX);
      const __bf16* hrow = s_h + (size_t)m * KFC;
      const float* wr = Wa + (size_t)j * HID;
      float s = ba[j];
      for (int k2 = 0; k2 < HID; ++k2) s += (float)hrow[k2] * wr[k2];
      float v = fexp(s);
      size_t rec = ((size_t)(brow0 + m) * TT + t) * KMIX;
      if (j < KMIX) {
        s_alp[m * KMIX + j] = v;  out[OFF_ALP + rec + j] = v;
      } else if (j < 2 * KMIX) {
        s_bet[m * KMIX + j - KMIX] = v;  out[OFF_BET + rec + (j - KMIX)] = v;
      } else {
        int kk = j - 2 * KMIX;
        float kap = v * 0.2f + s_kap[m * KMIX + kk];
        s_kap[m * KMIX + kk] = kap;  out[OFF_KAP + rec + kk] = kap;
      }
    }
    __syncthreads();

    // ---- phi[m][u] -> d_out (also reused below through L2) ----
    for (int i = tid; i < 16 * UU; i += 512) {
      int m = i >> 10, u = i & (UU - 1);
      float fu = (float)u, s = 0.f;
#pragma unroll
      for (int kk = 0; kk < KMIX; ++kk) {
        float d = s_kap[m * KMIX + kk] - fu;
        s += s_alp[m * KMIX + kk] * fexp(-s_bet[m * KMIX + kk] * d * d);
      }
      out[OFF_PHI + ((size_t)(brow0 + m) * TT + t) * UU + u] = s;
    }
    __threadfence();
    __syncthreads();

    // ---- w[m][e] = sum_u phi[m][u] * context[b_m][u][e] ----
    for (int i = tid; i < 16 * EE; i += 512) {
      int m = i >> 6, e = i & 63;
      const float* ph = out + OFF_PHI + ((size_t)(brow0 + m) * TT + t) * UU;
      const float* cx = context + (size_t)(brow0 + m) * UU * EE + e;
      float s = 0.f;
      for (int u = 0; u < UU; ++u) s += ph[u] * cx[(size_t)u * EE];
      s_wf[i] = s;
      s_xw[m * KXW + 3 + e] = (__bf16)s;   // new w feeds LSTM1/2
    }
    __syncthreads();

    // ---- LSTM1 gates: [x|w] @ Wxw1 + h0 @ Wh01 + h1_prev @ Whh1 ----
    init_gates(acc, b1, ntile, lane);
    gemm_acc<KXW / 32>(acc, s_xw, KXW, ws + O_WXW1, KXW, ntile, lane);
    gemm_acc<HID / 32>(acc, s_h,        KFC, ws + O_WH01, HID, ntile, lane);
    gemm_acc<HID / 32>(acc, s_h + HID,  KFC, ws + O_WHH1, HID, ntile, lane);
    __syncthreads();
    lstm_update(acc, c1st, s_h, HID, n0, lane);
    __syncthreads();

    // ---- LSTM2 gates: [x|w] @ Wxw2 + h1 @ Wh12 + h2_prev @ Whh2 ----
    init_gates(acc, b2, ntile, lane);
    gemm_acc<KXW / 32>(acc, s_xw, KXW, ws + O_WXW2, KXW, ntile, lane);
    gemm_acc<HID / 32>(acc, s_h + HID,     KFC, ws + O_WH12, HID, ntile, lane);
    gemm_acc<HID / 32>(acc, s_h + 2 * HID, KFC, ws + O_WHH2, HID, ntile, lane);
    __syncthreads();
    lstm_update(acc, c2st, s_h, 2 * HID, n0, lane);
    __syncthreads();

    // ---- output projection: [h0|h1|h2] @ Wfc.T (+bfc), waves 0..7 ----
    if (wv < 8) {
      int ncol = wv * 16;
      int c = ncol + (lane & 15), hi = lane >> 4;
      v8f oacc;
      {
        float bv = (c < NOUT) ? bfc[c] : 0.f;
#pragma unroll
        for (int r = 0; r < 8; ++r) oacc[r] = bv;
      }
      const __bf16* arow = s_h + (size_t)(lane & 15) * KFC;
#pragma unroll 1
      for (int kb = 0; kb < KFC / 32; ++kb) {
        v16bf a = load_a(arow, kb * 32, hi);
        v16bf b = load_b(ws + O_WFC, KFC, ncol, kb * 32, lane);
        oacc = WMMA_BF16(a, b, oacc);
      }
      if (c < NOUT) {
#pragma unroll
        for (int r = 0; r < 8; ++r) {
          int m = r + hi * 8;
          out[OFF_OUT + ((size_t)(brow0 + m) * TT + t) * NOUT + c] = oacc[r];
        }
      }
    }
    __syncthreads();
  }

  // ---- final carries ----
  for (int i = tid; i < 16 * HID; i += 512) {
    int m = i >> 9, k2 = i & (HID - 1);
    out[OFF_H0 + (size_t)(brow0 + m) * HID + k2] = (float)s_h[(size_t)m * KFC + k2];
    out[OFF_H1 + (size_t)(brow0 + m) * HID + k2] = (float)s_h[(size_t)m * KFC + HID + k2];
    out[OFF_H2 + (size_t)(brow0 + m) * HID + k2] = (float)s_h[(size_t)m * KFC + 2 * HID + k2];
  }
  {
    int hi = lane >> 4;
#pragma unroll
    for (int j = 0; j < 2; ++j) {
      int c = n0 + (lane & 15) + j * 16;
#pragma unroll
      for (int r = 0; r < 8; ++r) {
        int m = r + hi * 8;
        out[OFF_C0 + (size_t)(brow0 + m) * HID + c] = c0st[j][r];
        out[OFF_C1 + (size_t)(brow0 + m) * HID + c] = c1st[j][r];
        out[OFF_C2 + (size_t)(brow0 + m) * HID + c] = c2st[j][r];
      }
    }
  }
  for (int i = tid; i < 16 * EE; i += 512) {
    int m = i >> 6, e = i & 63;
    out[OFF_W + (size_t)(brow0 + m) * EE + e] = s_wf[i];
  }
  if (tid < 16 * KMIX) {
    int m = tid / KMIX, kk = tid - m * KMIX;
    out[OFF_KC + (size_t)(brow0 + m) * KMIX + kk] = s_kap[tid];
  }
}

// ---------------------------------------------------------------------------
extern "C" void kernel_launch(void* const* d_in, const int* in_sizes, int n_in,
                              void* d_out, int out_size, void* d_ws,
                              size_t ws_size, hipStream_t stream) {
  (void)in_sizes; (void)n_in; (void)out_size; (void)ws_size;
  const float* strokes = (const float*)d_in[0];
  const float* context = (const float*)d_in[1];
  const float* W_ih0 = (const float*)d_in[3];
  const float* W_hh0 = (const float*)d_in[4];
  const float* b0    = (const float*)d_in[5];
  const float* W_ih1 = (const float*)d_in[6];
  const float* W_hh1 = (const float*)d_in[7];
  const float* b1    = (const float*)d_in[8];
  const float* W_ih2 = (const float*)d_in[9];
  const float* W_hh2 = (const float*)d_in[10];
  const float* b2    = (const float*)d_in[11];
  const float* Wa    = (const float*)d_in[12];
  const float* ba    = (const float*)d_in[13];
  const float* Wfc   = (const float*)d_in[14];
  const float* bfc   = (const float*)d_in[15];
  __bf16* ws = (__bf16*)d_ws;
  float* out = (float*)d_out;

  auto blocks = [](size_t n) { return dim3((unsigned)((n + 255) / 256)); };
  dim3 thr(256);

  // zero-fill workspace (covers all K/N padding)
  k_zero_bf16<<<blocks(WS_TOT), thr, 0, stream>>>(ws, (int)WS_TOT);

  // bf16 weight staging, kept [N rows, K cols] row-major for B-fragment loads
  k_cvt<<<blocks(2048 * 67), thr, 0, stream>>>(W_ih0, ws + O_WXW0, 2048, 67, 0, 67, KXW, 0);
  k_cvt<<<blocks(2048 * 512), thr, 0, stream>>>(W_hh0, ws + O_WHH0, 2048, 512, 0, 512, HID, 0);

  k_cvt<<<blocks(2048 * 3),   thr, 0, stream>>>(W_ih1, ws + O_WXW1, 2048, 579, 0,   3,  KXW, 0);
  k_cvt<<<blocks(2048 * 64),  thr, 0, stream>>>(W_ih1, ws + O_WXW1, 2048, 579, 515, 64, KXW, 3);
  k_cvt<<<blocks(2048 * 512), thr, 0, stream>>>(W_ih1, ws + O_WH01, 2048, 579, 3, 512, HID, 0);
  k_cvt<<<blocks(2048 * 512), thr, 0, stream>>>(W_hh1, ws + O_WHH1, 2048, 512, 0, 512, HID, 0);

  k_cvt<<<blocks(2048 * 3),   thr, 0, stream>>>(W_ih2, ws + O_WXW2, 2048, 579, 0,   3,  KXW, 0);
  k_cvt<<<blocks(2048 * 64),  thr, 0, stream>>>(W_ih2, ws + O_WXW2, 2048, 579, 515, 64, KXW, 3);
  k_cvt<<<blocks(2048 * 512), thr, 0, stream>>>(W_ih2, ws + O_WH12, 2048, 579, 3, 512, HID, 0);
  k_cvt<<<blocks(2048 * 512), thr, 0, stream>>>(W_hh2, ws + O_WHH2, 2048, 512, 0, 512, HID, 0);

  k_cvt<<<blocks((size_t)NOUT * KFC), thr, 0, stream>>>(Wfc, ws + O_WFC, NOUT, KFC, 0, KFC, KFC, 0);

  // persistent recurrent kernel: 16 batch tiles x 16 waves
  k_rnn<<<dim3(16), dim3(512), 0, stream>>>(strokes, context, b0, b1, b2, Wa,
                                            ba, bfc, ws, out);
}
